// Glimpse_45140106281528
// MI455X (gfx1250) — matrix operand order repaired and checked by reading, not compiled, not run on hardware
//
#include <hip/hip_runtime.h>
#include <hip/hip_fp16.h>

typedef __attribute__((ext_vector_type(16))) _Float16 v16h;
typedef __attribute__((ext_vector_type(8)))  float    v8f;

#define BS 128
#define H  512
#define N  1024

// ---------------------------------------------------------------------------
// Pack W_ref (f32, [H][H] row-major, out=row, in=col) into f16 WMMA A-layout.
// A-tile = 16(M) x 32(K), contiguous K order (matches B staging):
//   lane<16 : M=lane,    element e: K=e
//   lane>=16: M=lane-16, element e: K=16+e
// Packed index: (((mt*16 + kt)*32 + lane)*16 + e) halves -> each lane loads
// 32 contiguous bytes; a wave reads 1KB contiguous.  mt=0..31, kt=0..15.
// ---------------------------------------------------------------------------
__global__ __launch_bounds__(256) void pack_w(const float* __restrict__ Wref,
                                              _Float16* __restrict__ Apack) {
    int tid  = blockIdx.x * 256 + threadIdx.x;      // 0 .. 512*512-1
    int e    = tid & 15;
    int ln   = (tid >> 4) & 31;
    int tile = tid >> 9;
    int kt   = tile & 15;
    int mt   = tile >> 4;
    int v    = e >> 1;
    int hlf  = e & 1;
    int kl   = 2 * v + hlf + ((ln >= 16) ? 16 : 0);
    int m    = mt * 16 + (ln & 15);
    int k    = kt * 32 + kl;
    Apack[tid] = (_Float16)Wref[m * H + k];
}

// ---------------------------------------------------------------------------
// out[b,h] = dot(W[h,:], X[b,:]) + bias[h]   (used for u1 and for final d)
// ---------------------------------------------------------------------------
__global__ __launch_bounds__(256) void gemv_bias(const float* __restrict__ X,
                                                 const float* __restrict__ W,
                                                 const float* __restrict__ bias,
                                                 float* __restrict__ out) {
    __shared__ float xs[H];
    int b = blockIdx.x, t = threadIdx.x;
    for (int i = t; i < H; i += 256) xs[i] = X[b * H + i];
    __syncthreads();
    for (int h = t; h < H; h += 256) {
        const float4* wr = (const float4*)(W + (size_t)h * H);
        float acc = 0.f;
        #pragma unroll 8
        for (int i = 0; i < H / 4; ++i) {
            float4 w4 = wr[i];
            acc += w4.x * xs[4 * i] + w4.y * xs[4 * i + 1] +
                   w4.z * xs[4 * i + 2] + w4.w * xs[4 * i + 3];
        }
        out[b * H + h] = acc + bias[h];
    }
}

// ---------------------------------------------------------------------------
// Main fused kernel: for block (b, n-strip of 32):
//   u2[:,strip] = W_ref @ ref[b][:,strip]   via v_wmma_f32_16x16x32_f16
//   s[b,n]      = sum_h v[h] * tanh(u1[b,h] + u2[h,n] + b_ref[h])
// 8 waves: wave w -> n-sub (w&1), m-tiles (w>>1)*8 .. +7  (8 v8f accums).
// B is double-buffered in LDS, stored transposed (row=n) so each lane reads
// a contiguous, 32B-aligned v16h in this code's contiguous K order, the same as A
// (lanes 0-15: K=0..15, lanes 16-31: K=16..31). One barrier per K-step;
// next-tile staging is issued right after the barrier (other buffer), then
// all 8 A tiles are preloaded into registers (16 b128 loads in flight)
// before the 8 back-to-back WMMAs, hiding L2 latency inside the wave.
// ---------------------------------------------------------------------------
__global__ __launch_bounds__(256) void fused_gemm_score(
        const float*    __restrict__ ref,
        const _Float16* __restrict__ Apack,
        const float*    __restrict__ u1,
        const float*    __restrict__ vvec,
        const float*    __restrict__ b_ref,
        float*          __restrict__ s) {
    const int b    = blockIdx.y;
    const int n0   = blockIdx.x * 32;
    const int t    = threadIdx.x;
    const int w    = t >> 5;
    const int ln   = t & 31;
    const int ln15 = ln & 15;
    const int hi   = ln >> 4;           // 0/1: lane half
    const int ns   = w & 1;             // n-subtile (16 cols)
    const int mtb  = (w >> 1) * 8;      // first m-tile of this wave

    __shared__ __align__(32) _Float16 Bs[2][32][48];   // [buf][n][k], padded
    __shared__ float sred[32];

    if (t < 32) sred[t] = 0.0f;

    v8f acc[8] = {};

    const float* refb = ref + (size_t)b * H * N;

    // staging role of this thread: one float4 along n per K-step
    const int sk = t >> 3;         // k row 0..31
    const int sn = (t & 7) * 4;    // n start 0,4,...,28

    auto stage = [&](int kt, int buf) {
        float4 r4 = *(const float4*)(refb + (size_t)(kt * 32 + sk) * N + n0 + sn);
        Bs[buf][sn + 0][sk] = (_Float16)r4.x;
        Bs[buf][sn + 1][sk] = (_Float16)r4.y;
        Bs[buf][sn + 2][sk] = (_Float16)r4.z;
        Bs[buf][sn + 3][sk] = (_Float16)r4.w;
    };

    // per-wave A base: tiles (mtb+m, kt) live at ((mt*16+kt)*32+ln)*16 halves
    const _Float16* abase = Apack + ((size_t)(mtb * 16) * 32 + ln) * 16;

    stage(0, 0);
    for (int kt = 0; kt < 16; ++kt) {
        __syncthreads();               // staging of buf (kt&1) complete;
                                       // all reads of buf (kt&1)^1 complete
        const int buf = kt & 1;
        if (kt < 15) stage(kt + 1, buf ^ 1);   // prefetch next tile (no race)

        const v16h bvec = *(const v16h*)(&Bs[buf][ns * 16 + ln15][hi * 16]);

        v16h av[8];
        #pragma unroll
        for (int m = 0; m < 8; ++m)    // 16 independent b128 loads in flight
            av[m] = *(const v16h*)(abase + ((size_t)(m * 16 + kt) * 32) * 16);

        #pragma unroll
        for (int m = 0; m < 8; ++m)    // back-to-back WMMAs, shared B
            acc[m] = __builtin_amdgcn_wmma_f32_16x16x32_f16(
                false, av[m], false, bvec, (short)0, acc[m], false, false);
    }

    // Epilogue: C layout -> lane ln, VGPR r holds (h = mtile*16 + r + 8*hi,
    // n = n0 + ns*16 + ln15).  Lanes ln and ln+16 share n with disjoint h.
    float local = 0.0f;
    #pragma unroll
    for (int m = 0; m < 8; ++m) {
        int hbase = (mtb + m) * 16 + hi * 8;
        #pragma unroll
        for (int r = 0; r < 8; ++r) {
            int h  = hbase + r;
            float u2 = acc[m][r] + b_ref[h];
            local += vvec[h] * tanhf(u1[b * H + h] + u2);
        }
    }
    atomicAdd(&sred[ns * 16 + ln15], local);   // ds_add_f32
    __syncthreads();
    if (t < 32) s[b * N + n0 + t] = sred[t];
}

// ---------------------------------------------------------------------------
// Masked softmax over n:  a = softmax(s - 1e8*mask)
// ---------------------------------------------------------------------------
__global__ __launch_bounds__(256) void softmax_mask(const float* __restrict__ s,
                                                    const int*   __restrict__ mask,
                                                    float*       __restrict__ a) {
    __shared__ float red[256];
    int b = blockIdx.x, t = threadIdx.x;
    float uv[4];
    float mx = -3.4e38f;
    #pragma unroll
    for (int i = 0; i < 4; ++i) {
        int n = t + i * 256;
        float u = s[b * N + n] - 1e8f * (float)mask[b * N + n];
        uv[i] = u;
        mx = fmaxf(mx, u);
    }
    red[t] = mx;
    __syncthreads();
    for (int off = 128; off; off >>= 1) {
        if (t < off) red[t] = fmaxf(red[t], red[t + off]);
        __syncthreads();
    }
    mx = red[0];
    __syncthreads();
    float sum = 0.f;
    #pragma unroll
    for (int i = 0; i < 4; ++i) { uv[i] = expf(uv[i] - mx); sum += uv[i]; }
    red[t] = sum;
    __syncthreads();
    for (int off = 128; off; off >>= 1) {
        if (t < off) red[t] += red[t + off];
        __syncthreads();
    }
    float inv = 1.0f / red[0];
    #pragma unroll
    for (int i = 0; i < 4; ++i) a[b * N + t + i * 256] = uv[i] * inv;
}

// ---------------------------------------------------------------------------
// rbar[b,k] = sum_n a[b,n] * ref[b,k,n]   (pure-bandwidth second pass)
// ---------------------------------------------------------------------------
__global__ __launch_bounds__(256) void weighted_ref(const float* __restrict__ ref,
                                                    const float* __restrict__ a,
                                                    float* __restrict__ rbar) {
    __shared__ float as[N];
    int b = blockIdx.x, t = threadIdx.x;
    for (int i = t; i < N; i += 256) as[i] = a[b * N + i];
    __syncthreads();
    int w = t >> 5, ln = t & 31;
    const float* refb = ref + (size_t)b * H * N;
    for (int k = w; k < H; k += 8) {
        const float4* row = (const float4*)(refb + (size_t)k * N);
        float sum = 0.f;
        #pragma unroll
        for (int it = 0; it < N / 128; ++it) {
            int nq = it * 32 + ln;
            float4 r4 = row[nq];
            sum += r4.x * as[nq * 4] + r4.y * as[nq * 4 + 1] +
                   r4.z * as[nq * 4 + 2] + r4.w * as[nq * 4 + 3];
        }
        #pragma unroll
        for (int off = 16; off; off >>= 1) sum += __shfl_down(sum, off, 32);
        if (ln == 0) rbar[b * H + k] = sum;
    }
}

// ---------------------------------------------------------------------------
extern "C" void kernel_launch(void* const* d_in, const int* in_sizes, int n_in,
                              void* d_out, int out_size, void* d_ws, size_t ws_size,
                              hipStream_t stream) {
    const float* q     = (const float*)d_in[0];
    const float* ref   = (const float*)d_in[1];
    const int*   mask  = (const int*)  d_in[2];
    const float* v     = (const float*)d_in[3];
    const float* W_q   = (const float*)d_in[4];
    const float* b_q   = (const float*)d_in[5];
    const float* W_ref = (const float*)d_in[6];
    const float* b_ref = (const float*)d_in[7];
    float* out = (float*)d_out;

    char* w = (char*)d_ws;
    _Float16* Apack = (_Float16*)w;                                   // 512 KB
    float* u1   = (float*)(w + 524288);                               // 256 KB
    float* s    = (float*)(w + 524288 + 262144);                      // 512 KB
    float* a    = (float*)(w + 524288 + 262144 + 524288);             // 512 KB
    float* rbar = (float*)(w + 524288 + 262144 + 524288 + 524288);    // 256 KB

    pack_w<<<(H * H) / 256, 256, 0, stream>>>(W_ref, Apack);
    gemv_bias<<<BS, 256, 0, stream>>>(q, W_q, b_q, u1);               // u1
    dim3 g1(N / 32, BS);
    fused_gemm_score<<<g1, 256, 0, stream>>>(ref, Apack, u1, v, b_ref, s);
    softmax_mask<<<BS, 256, 0, stream>>>(s, mask, a);
    weighted_ref<<<BS, 256, 0, stream>>>(ref, a, rbar);
    gemv_bias<<<BS, 256, 0, stream>>>(rbar, W_ref, b_ref, out);       // d
}
